// SAR_63385127354414
// MI455X (gfx1250) — matrix operand order repaired and checked by reading, not compiled, not run on hardware
//
#include <hip/hip_runtime.h>
#include <stdint.h>
#include <stddef.h>

// ---------------- CDNA5 (gfx1250) wave32 WMMA types ----------------
typedef __attribute__((ext_vector_type(16))) _Float16 v16h;
typedef __attribute__((ext_vector_type(8)))  _Float16 v8h;
typedef __attribute__((ext_vector_type(8)))  float    v8f;

#define NBATCH 4
#define CH     256
#define TT     1728          // 12*12*12
#define MROWS  (NBATCH*TT)   // 6912
#define NHEAD  4
#define HDIM   64
#define CT     (CH*TT)       // 442368 per batch

__device__ __constant__ float kLamInit  = 0.55605820417f;   // 0.8 - 0.6*exp(-0.9)
__device__ __constant__ float kOneMinus = 0.44394179583f;

__device__ __forceinline__ v8f wmma_f16(v16h a, v16h b, v8f c) {
  // D = A(16x32 f16) x B(32x16 f16) + C(16x16 f32)  -> v_wmma_f32_16x16x32_f16
  return __builtin_amdgcn_wmma_f32_16x16x32_f16(false, a, false, b, (short)0, c,
                                                false, false);
}

// A fragment, 16x32 f16, row-major source (ld in halves).
// lane L: m=L&15, kb=(L&16)?8:0 ; halves 0..7 -> K=kb..kb+7 ; 8..15 -> K=16+kb..
__device__ __forceinline__ v16h load_afrag(const _Float16* base, int ld) {
  int lane = threadIdx.x & 31;
  int m  = lane & 15;
  int kb = (lane & 16) ? 8 : 0;
  const _Float16* p = base + (size_t)m * ld;
  v8h lo = *(const v8h*)(p + kb);
  v8h hi = *(const v8h*)(p + 16 + kb);
  v16h r;
#pragma unroll
  for (int i = 0; i < 8; i++) { r[i] = lo[i]; r[i + 8] = hi[i]; }
  return r;
}

// B fragment, 32x16 f16 where B = S^T : B[k][n] = S[nrow+n][kcol+k]
// lane L: n=L&15, kb=(L&16)?8:0 ; halves 0..7 -> K=kb..kb+7 ; 8..15 -> K=16+kb..
__device__ __forceinline__ v16h load_bfrag(const _Float16* S, int ld, int nrow, int kcol) {
  int lane = threadIdx.x & 31;
  const _Float16* p = S + (size_t)(nrow + (lane & 15)) * ld + kcol + ((lane & 16) ? 8 : 0);
  v8h lo = *(const v8h*)(p);
  v8h hi = *(const v8h*)(p + 16);
  v16h r;
#pragma unroll
  for (int i = 0; i < 8; i++) { r[i] = lo[i]; r[i + 8] = hi[i]; }
  return r;
}

// row reductions inside each 16-lane half (C-frag rows live across 16 lanes)
__device__ __forceinline__ float halfred_max(float x) {
  x = fmaxf(x, __shfl_xor(x, 1, 32));
  x = fmaxf(x, __shfl_xor(x, 2, 32));
  x = fmaxf(x, __shfl_xor(x, 4, 32));
  x = fmaxf(x, __shfl_xor(x, 8, 32));
  return x;
}
__device__ __forceinline__ float halfred_sum(float x) {
  x += __shfl_xor(x, 1, 32);
  x += __shfl_xor(x, 2, 32);
  x += __shfl_xor(x, 4, 32);
  x += __shfl_xor(x, 8, 32);
  return x;
}

// ---------------- K0: gamma sum + per-head lambda ----------------
__global__ void prep_kernel(const float* __restrict__ gamma,
                            const float* __restrict__ lq1, const float* __restrict__ lk1,
                            const float* __restrict__ lq2, const float* __restrict__ lk2,
                            float* __restrict__ misc) {  // misc[0]=1/sum(gamma), misc[1..4]=lam
  int lane = threadIdx.x;
  if (lane == 4) {
    float s = 0.f;
    for (int i = 0; i < CH; i++) s += gamma[i];
    misc[0] = 1.0f / s;
  }
  if (lane < 4) {
    float d1 = 0.f, d2 = 0.f;
    for (int i = 0; i < HDIM; i++) {
      d1 += lq1[lane * HDIM + i] * lk1[lane * HDIM + i];
      d2 += lq2[lane * HDIM + i] * lk2[lane * HDIM + i];
    }
    misc[1 + lane] = __expf(d1) - __expf(d2) + kLamInit;
  }
}

// ---------------- K1: convert weights to f16 ----------------
// wAll rows: [0,256)=wq1 [256,512)=wq2 [512,768)=wk1 [768,1024)=wk2 [1024,1536)=wv
__global__ void convw_kernel(const float* __restrict__ wq1, const float* __restrict__ wq2,
                             const float* __restrict__ wk1, const float* __restrict__ wk2,
                             const float* __restrict__ wv,  const float* __restrict__ wc,
                             _Float16* __restrict__ wAll, _Float16* __restrict__ wc16) {
  int idx = blockIdx.x * blockDim.x + threadIdx.x;
  const int NW = 1536 * 256;
  if (idx < NW) {
    int row = idx >> 8, col = idx & 255;
    const float* src; int r;
    if (row < 256)       { src = wq1; r = row; }
    else if (row < 512)  { src = wq2; r = row - 256; }
    else if (row < 768)  { src = wk1; r = row - 512; }
    else if (row < 1024) { src = wk2; r = row - 768; }
    else                 { src = wv;  r = row - 1024; }
    wAll[idx] = (_Float16)src[r * 256 + col];
  } else if (idx < NW + 256 * 512) {
    int k = idx - NW;
    wc16[k] = (_Float16)wc[k];
  }
}

// ---------------- K2: groupnorm stats (64 groups of 27648) ----------------
__global__ __launch_bounds__(256) void gn_stats_kernel(const float* __restrict__ x,
                                                       float* __restrict__ stats) {
  int blk = blockIdx.x;  // b*16 + g
  const float* p = x + (size_t)blk * 27648;
  float s = 0.f, s2 = 0.f;
  for (int i = threadIdx.x; i < 27648; i += 256) { float v = p[i]; s += v; s2 += v * v; }
  for (int o = 16; o > 0; o >>= 1) { s += __shfl_xor(s, o, 32); s2 += __shfl_xor(s2, o, 32); }
  __shared__ float rs[8], rq[8];
  int wave = threadIdx.x >> 5, lane = threadIdx.x & 31;
  if (lane == 0) { rs[wave] = s; rq[wave] = s2; }
  __syncthreads();
  if (threadIdx.x == 0) {
    float S = 0.f, S2 = 0.f;
    for (int w = 0; w < 8; w++) { S += rs[w]; S2 += rq[w]; }
    const float n = 27648.0f;
    float mean = S / n;
    float var = fmaxf((S2 - n * mean * mean) / (n - 1.0f), 0.0f);  // ddof=1
    stats[blk * 2]     = mean;
    stats[blk * 2 + 1] = 1.0f / (sqrtf(var) + 1e-10f);
  }
}

// ---------------- K3: gate: x1(f16), x2(f32) ----------------
__global__ void gate_kernel(const float* __restrict__ x, const float* __restrict__ gamma,
                            const float* __restrict__ stats, const float* __restrict__ misc,
                            _Float16* __restrict__ x1, float* __restrict__ x2) {
  int idx = blockIdx.x * blockDim.x + threadIdx.x;
  if (idx >= NBATCH * CT) return;
  int b = idx / CT, r = idx % CT, c = r / TT;
  int blk = b * 16 + (c >> 4);
  float mean = stats[blk * 2], inv = stats[blk * 2 + 1];
  float gm = gamma[c];
  float xv = x[idx];
  float gn = (xv - mean) * inv * gm;
  float z = gn * gm * misc[0];              // gn * gamma/sum(gamma)
  bool info = (z >= 0.0f);                  // sigmoid(z) >= 0.5
  x1[idx] = (_Float16)(info ? xv : 0.0f);
  x2[idx] = info ? 0.0f : xv;
}

// ---------------- K4: projection GEMM (WMMA) ----------------
// A = x1 (6912,256) f16 ; B rows from wAll (1536,256) ; block = 64x64 tile, 4 waves
// Q1/Q2 outputs are pre-scaled by 1/sqrt(HDIM)=0.125 (exact power of 2 in f16),
// so the attention hot loop needs no per-element logit scaling.
__global__ __launch_bounds__(128) void proj_gemm_kernel(
    const _Float16* __restrict__ A, const _Float16* __restrict__ wAll,
    _Float16* __restrict__ Q1, _Float16* __restrict__ Q2,
    _Float16* __restrict__ K1, _Float16* __restrict__ K2,
    _Float16* __restrict__ Vt) {
  int wave = threadIdx.x >> 5, lane = threadIdx.x & 31;
  int rb = blockIdx.x * 64 + wave * 16;
  int cb = blockIdx.y * 64;
  v8f acc[4] = {};
  const _Float16* Arow = A + (size_t)rb * CH;
#pragma unroll 2
  for (int kc = 0; kc < CH; kc += 32) {
    v16h af = load_afrag(Arow + kc, CH);
#pragma unroll
    for (int j = 0; j < 4; j++) {
      v16h bf = load_bfrag(wAll, CH, cb + 16 * j, kc);
      acc[j] = wmma_f16(af, bf, acc[j]);
    }
  }
  int nn = lane & 15, mh = (lane >> 4) * 8;
  if (cb < 1024) {  // Q1/Q2/K1/K2 row-major (MROWS,256)
    _Float16* dst = (cb < 256) ? Q1 : (cb < 512) ? Q2 : (cb < 768) ? K1 : K2;
    float scl = (cb < 512) ? 0.125f : 1.0f;  // fold 1/sqrt(d) into Q
    int cbase = cb & 255;
#pragma unroll
    for (int j = 0; j < 4; j++)
#pragma unroll
      for (int m = 0; m < 8; m++)
        dst[(size_t)(rb + mh + m) * CH + cbase + 16 * j + nn] = (_Float16)(acc[j][m] * scl);
  } else {          // V stored transposed: Vt (NBATCH,512,TT)
    int b = rb / TT, t0 = rb % TT;
#pragma unroll
    for (int j = 0; j < 4; j++) {
      int e = (cb - 1024) + 16 * j + nn;
      v8h pk;
#pragma unroll
      for (int m = 0; m < 8; m++) pk[m] = (_Float16)acc[j][m];
      *(v8h*)(Vt + ((size_t)b * 512 + e) * TT + t0 + mh) = pk;  // 8 consecutive t
    }
  }
}

// ---------------- K5: fused differential flash attention (WMMA) ----------------
// grid (T/64, NHEAD, NBATCH), 4 waves: wave owns 16 q-rows, both softmax streams online.
__global__ __launch_bounds__(128) void attn_kernel(
    const _Float16* __restrict__ Q1, const _Float16* __restrict__ Q2,
    const _Float16* __restrict__ K1, const _Float16* __restrict__ K2,
    const _Float16* __restrict__ Vt, const float* __restrict__ lamArr,
    _Float16* __restrict__ Y) {
  __shared__ __align__(32) _Float16 pls[2][4][16 * 32];  // per-wave P tiles (no barriers)
  int wave = threadIdx.x >> 5, lane = threadIdx.x & 31;
  int h = blockIdx.y, b = blockIdx.z;
  int q0 = blockIdx.x * 64 + wave * 16;
  size_t g0 = (size_t)b * TT + q0;
  int nn = lane & 15, mh = (lane >> 4) * 8;

  const _Float16* Kb[2] = { K1 + (size_t)b * TT * CH, K2 + (size_t)b * TT * CH };
  const _Float16* Vb = Vt + (size_t)b * 512 * TT;
  v16h qf[2][2];
  qf[0][0] = load_afrag(Q1 + g0 * CH + h * HDIM,      CH);   // already *0.125
  qf[0][1] = load_afrag(Q1 + g0 * CH + h * HDIM + 32, CH);
  qf[1][0] = load_afrag(Q2 + g0 * CH + h * HDIM,      CH);
  qf[1][1] = load_afrag(Q2 + g0 * CH + h * HDIM + 32, CH);

  v8f acc[2][8] = {};
  float mrow[2][8], lrow[2][8];
#pragma unroll
  for (int st = 0; st < 2; st++)
#pragma unroll
    for (int m = 0; m < 8; m++) { mrow[st][m] = -1e30f; lrow[st][m] = 0.f; }

  int nk = (q0 + 47) >> 5;  // causal: key tiles of 32 covering keys <= q0+15
  for (int ki = 0; ki < nk; ki++) {
    int kc = ki * 32;
    // prefetch next tile's K rows and every 4th V row into cache (L2-resident stream)
    if (ki + 1 < nk) {
      int kn = kc + 32;
      __builtin_prefetch(Kb[0] + (size_t)(kn + lane) * CH + h * HDIM, 0, 1);
      __builtin_prefetch(Kb[1] + (size_t)(kn + lane) * CH + h * HDIM, 0, 1);
      __builtin_prefetch(Vb + (size_t)(h * 128 + lane * 4) * TT + kn, 0, 1);
    }
    v8f sfr[2][2];
#pragma unroll
    for (int st = 0; st < 2; st++)
#pragma unroll
      for (int sub = 0; sub < 2; sub++) {
        v8f s = {};
#pragma unroll
        for (int dch = 0; dch < 2; dch++) {
          v16h bk = load_bfrag(Kb[st], CH, kc + 16 * sub, h * HDIM + 32 * dch);
          s = wmma_f16(qf[st][dch], bk, s);
        }
        sfr[st][sub] = s;
      }
    // causal mask only on diagonal-crossing tiles (uniform branch per wave)
    if (kc + 32 > q0) {
#pragma unroll
      for (int st = 0; st < 2; st++)
#pragma unroll
        for (int sub = 0; sub < 2; sub++)
#pragma unroll
          for (int m = 0; m < 8; m++) {
            int qi = q0 + mh + m;
            int kx = kc + 16 * sub + nn;
            if (kx > qi) sfr[st][sub][m] = -1e30f;
          }
    }
    // dual-stream online softmax + P -> LDS (A-layout round trip)
#pragma unroll
    for (int st = 0; st < 2; st++) {
      float corr[8];
#pragma unroll
      for (int m = 0; m < 8; m++) {
        float tm = halfred_max(fmaxf(sfr[st][0][m], sfr[st][1][m]));
        float mo = mrow[st][m];
        float mn = fmaxf(mo, tm);
        float c0 = __expf(mo - mn);
        float p0 = __expf(sfr[st][0][m] - mn);
        float p1 = __expf(sfr[st][1][m] - mn);
        sfr[st][0][m] = p0; sfr[st][1][m] = p1;
        lrow[st][m] = lrow[st][m] * c0 + halfred_sum(p0 + p1);
        mrow[st][m] = mn;
        corr[m] = c0;
      }
#pragma unroll
      for (int j = 0; j < 8; j++)
#pragma unroll
        for (int m = 0; m < 8; m++) acc[st][j][m] *= corr[m];
      _Float16* pl = pls[st][wave];
#pragma unroll
      for (int sub = 0; sub < 2; sub++)
#pragma unroll
        for (int m = 0; m < 8; m++)
          pl[(mh + m) * 32 + 16 * sub + nn] = (_Float16)sfr[st][sub][m];
    }
    v16h pa0 = load_afrag(pls[0][wave], 32);
    v16h pa1 = load_afrag(pls[1][wave], 32);
#pragma unroll
    for (int j = 0; j < 8; j++) {
      v16h bv = load_bfrag(Vb, TT, h * 128 + 16 * j, kc);  // shared by both streams
      acc[0][j] = wmma_f16(pa0, bv, acc[0][j]);
      acc[1][j] = wmma_f16(pa1, bv, acc[1][j]);
    }
  }

  // finalize: y = o1/l1 - lam*o2/l2, LN over 128, *(1-lam_init)
  float lam = lamArr[h];
  float r1[8], r2[8], sm[8], sq[8];
#pragma unroll
  for (int m = 0; m < 8; m++) {
    r1[m] = 1.0f / lrow[0][m];
    r2[m] = lam / lrow[1][m];
    sm[m] = 0.f; sq[m] = 0.f;
  }
#pragma unroll
  for (int j = 0; j < 8; j++)
#pragma unroll
    for (int m = 0; m < 8; m++) {
      float y = acc[0][j][m] * r1[m] - acc[1][j][m] * r2[m];
      acc[0][j][m] = y;
      sm[m] += y; sq[m] += y * y;
    }
  float mu[8], sc[8];
#pragma unroll
  for (int m = 0; m < 8; m++) {
    float tot  = halfred_sum(sm[m]);
    float tot2 = halfred_sum(sq[m]);
    float mean = tot * (1.0f / 128.0f);
    float var  = tot2 * (1.0f / 128.0f) - mean * mean;
    mu[m] = mean;
    sc[m] = rsqrtf(fmaxf(var, 0.f) + 1e-5f) * kOneMinus;
  }
#pragma unroll
  for (int j = 0; j < 8; j++)
#pragma unroll
    for (int m = 0; m < 8; m++)
      Y[(g0 + mh + m) * 512 + h * 128 + 16 * j + nn] =
          (_Float16)((acc[0][j][m] - mu[m]) * sc[m]);
}

// ---------------- K6: output GEMM + fused channel-half swap add ----------------
// out[b, c, s] = (Y @ wc^T)[b, t, c2] + x2[b, (c+128)%256, s], all on the same flat buffer.
__global__ __launch_bounds__(128) void out_gemm_kernel(const _Float16* __restrict__ Y,
                                                       const _Float16* __restrict__ wc16,
                                                       const float* __restrict__ x2,
                                                       float* __restrict__ out) {
  int wave = threadIdx.x >> 5, lane = threadIdx.x & 31;
  int rb = blockIdx.x * 64 + wave * 16;
  int cb = blockIdx.y * 64;
  v8f acc[4] = {};
  const _Float16* Arow = Y + (size_t)rb * 512;
#pragma unroll 2
  for (int kc = 0; kc < 512; kc += 32) {
    v16h af = load_afrag(Arow + kc, 512);
#pragma unroll
    for (int j = 0; j < 4; j++) {
      v16h bf = load_bfrag(wc16, 512, cb + 16 * j, kc);
      acc[j] = wmma_f16(af, bf, acc[j]);
    }
  }
  int nn = lane & 15, mh = (lane >> 4) * 8;
  int b = rb / TT, t0 = rb % TT;
  const float* x2b = x2 + (size_t)b * CT;
  float* outb = out + (size_t)b * CT;
#pragma unroll
  for (int j = 0; j < 4; j++)
#pragma unroll
    for (int m = 0; m < 8; m++) {
      int f = (t0 + mh + m) * CH + cb + 16 * j + nn;  // flat within batch (t*256+c2 view)
      int c = f / TT;                                  // channel in (C,H,W,D) view
      int s = f - c * TT;
      int c2 = (c + 128) & 255;
      outb[f] = acc[j][m] + x2b[c2 * TT + s];
    }
}

// ---------------- launcher ----------------
extern "C" void kernel_launch(void* const* d_in, const int* in_sizes, int n_in,
                              void* d_out, int out_size, void* d_ws, size_t ws_size,
                              hipStream_t stream) {
  const float* x     = (const float*)d_in[0];
  const float* gamma = (const float*)d_in[1];
  // d_in[2] = beta (zeros) unused
  const float* wq1 = (const float*)d_in[3];
  const float* wq2 = (const float*)d_in[4];
  const float* wk1 = (const float*)d_in[5];
  const float* wk2 = (const float*)d_in[6];
  const float* wv  = (const float*)d_in[7];
  const float* wc  = (const float*)d_in[8];
  const float* lq1 = (const float*)d_in[9];
  const float* lk1 = (const float*)d_in[10];
  const float* lq2 = (const float*)d_in[11];
  const float* lk2 = (const float*)d_in[12];

  char* ws = (char*)d_ws;
  size_t off = 0;
  auto alloc = [&](size_t bytes) { size_t o = off; off += (bytes + 255) & ~(size_t)255; return o; };
  float*    stats = (float*)(ws + alloc(64 * 2 * sizeof(float)));
  float*    misc  = (float*)(ws + alloc(8 * sizeof(float)));          // [0]=1/sum(g), [1..4]=lam
  _Float16* x1    = (_Float16*)(ws + alloc((size_t)MROWS * CH * 2));
  float*    x2    = (float*)   (ws + alloc((size_t)MROWS * CH * 4));
  _Float16* wAll  = (_Float16*)(ws + alloc((size_t)1536 * 256 * 2));
  _Float16* wc16  = (_Float16*)(ws + alloc((size_t)256 * 512 * 2));
  _Float16* Q1    = (_Float16*)(ws + alloc((size_t)MROWS * CH * 2));
  _Float16* Q2    = (_Float16*)(ws + alloc((size_t)MROWS * CH * 2));
  _Float16* K1    = (_Float16*)(ws + alloc((size_t)MROWS * CH * 2));
  _Float16* K2    = (_Float16*)(ws + alloc((size_t)MROWS * CH * 2));
  _Float16* Vt    = (_Float16*)(ws + alloc((size_t)NBATCH * 512 * TT * 2));
  _Float16* Y     = (_Float16*)(ws + alloc((size_t)MROWS * 512 * 2));
  (void)ws_size;  // ~40 MB used

  prep_kernel<<<1, 32, 0, stream>>>(gamma, lq1, lk1, lq2, lk2, misc);
  convw_kernel<<<(1536 * 256 + 256 * 512 + 255) / 256, 256, 0, stream>>>(
      wq1, wq2, wk1, wk2, wv, wc, wAll, wc16);
  gn_stats_kernel<<<64, 256, 0, stream>>>(x, stats);
  gate_kernel<<<(NBATCH * CT + 255) / 256, 256, 0, stream>>>(x, gamma, stats, misc, x1, x2);
  proj_gemm_kernel<<<dim3(MROWS / 64, 1536 / 64), 128, 0, stream>>>(x1, wAll, Q1, Q2, K1, K2, Vt);
  attn_kernel<<<dim3(TT / 64, NHEAD, NBATCH), 128, 0, stream>>>(Q1, Q2, K1, K2, Vt, misc + 1, Y);
  out_gemm_kernel<<<dim3(MROWS / 64, CH / 64), 128, 0, stream>>>(Y, wc16, x2, (float*)d_out);
}
